// BertSelfAttention_86517821215406
// MI455X (gfx1250) — matrix operand order repaired and checked by reading, not compiled, not run on hardware
//
#include <hip/hip_runtime.h>
#include <hip/hip_bf16.h>

// ---------------------------------------------------------------------------
// BERT self-attention with relative-position bias, CDNA5 (gfx1250) wave32.
// All large contractions use v_wmma_f32_16x16x32_bf16 (bf16 in, f32 acc).
// LDS staging uses GLOBAL_LOAD_ASYNC_TO_LDS_B128 (ASYNCcnt) where the copy
// is byte-exact.
// ---------------------------------------------------------------------------

#define DEVINL __device__ __forceinline__

typedef __bf16 bf16_t;
typedef bf16_t bf16x16 __attribute__((ext_vector_type(16)));
typedef float  f32x8   __attribute__((ext_vector_type(8)));
typedef int    v4i_vs  __attribute__((__vector_size__(4 * sizeof(int))));

static constexpr int Bc = 2;     // batch
static constexpr int Sq = 1024;  // sequence
static constexpr int Hn = 12;    // heads
static constexpr int DH = 64;    // head dim
static constexpr int Dm = 768;   // hidden
static constexpr int BH = Bc * Hn;  // 24 batch*head rows

DEVINL f32x8 wmma_bf16(bf16x16 a, bf16x16 b, f32x8 c) {
  return __builtin_amdgcn_wmma_f32_16x16x32_bf16(false, a, false, b, (short)0,
                                                 c, false, false);
}

// ---- async global -> LDS copy (16B per lane call) -------------------------
// Builtin signature (leaked by clang diagnostic):
//   (v4i addrspace(1)* src, v4i addrspace(3)* dst, imm offset, imm cpol)
DEVINL void async_ld_b128(void* lds_dst, const void* gsrc) {
#if __has_builtin(__builtin_amdgcn_global_load_async_to_lds_b128)
  __builtin_amdgcn_global_load_async_to_lds_b128(
      (__attribute__((address_space(1))) v4i_vs*)(uintptr_t)gsrc,
      (__attribute__((address_space(3))) v4i_vs*)(uintptr_t)(uint32_t)(
          uintptr_t)lds_dst,
      0, 0);
#else
  const uint32_t lo = (uint32_t)(uintptr_t)lds_dst;  // low 32 bits = LDS addr
  asm volatile("global_load_async_to_lds_b128 %0, %1, off" ::"v"(lo),
               "v"((uint64_t)(uintptr_t)gsrc)
               : "memory");
#endif
}

DEVINL void wait_async0() {
#if __has_builtin(__builtin_amdgcn_s_wait_asynccnt)
  __builtin_amdgcn_s_wait_asynccnt(0);
#else
  asm volatile("s_wait_asynccnt 0" ::: "memory");
#endif
}

// A operand (16x32, row-major source): lane L holds row m=L&15.
// e0..e15 = K = kb+0..15,  kb = 16*(L>>4).
DEVINL bf16x16 load_a_f32(const float* row, int col0, int lane) {
  const int kb = ((lane >> 4) & 1) * 16;
  const float* p0 = row + col0 + kb;
  const float* p1 = row + col0 + 8 + kb;
  bf16x16 a;
#pragma unroll
  for (int i = 0; i < 8; ++i) { a[i] = (bf16_t)p0[i]; a[8 + i] = (bf16_t)p1[i]; }
  return a;
}

DEVINL bf16x16 load_a_bf16(const bf16_t* row, int col0, int lane) {
  const int kb = ((lane >> 4) & 1) * 16;
  const bf16_t* p0 = row + col0 + kb;
  const bf16_t* p1 = row + col0 + 8 + kb;
  bf16x16 a;
#pragma unroll
  for (int i = 0; i < 8; ++i) { a[i] = p0[i]; a[8 + i] = p1[i]; }
  return a;
}

// B operand (32x16): lane L holds column n=L&15, e_i = B[(L>>4)*16 + i][n].
DEVINL bf16x16 load_b_contig_bf16(const bf16_t* p) {
  bf16x16 b;
#pragma unroll
  for (int i = 0; i < 16; ++i) b[i] = p[i];
  return b;
}

DEVINL bf16x16 load_b_contig_f32_nt(const float* p) {
  bf16x16 b;
#pragma unroll
  for (int i = 0; i < 16; ++i) b[i] = (bf16_t)__builtin_nontemporal_load(p + i);
  return b;
}

// ---------------------------------------------------------------------------
// Kernel 1: projection  Y = X(2048x768) * W(768x768) + bias, bf16 output in
// head-major layout [bh][s][dh].  One wave per 16x16 output tile.
// ---------------------------------------------------------------------------
__global__ void proj_kernel(const float* __restrict__ X,
                            const float* __restrict__ W,
                            const float* __restrict__ bias,
                            bf16_t* __restrict__ Y) {
  const int lane = threadIdx.x & 31;
  const int nt = blockIdx.x, mt = blockIdx.y;
  const int col = nt * 16 + (lane & 15);
  const float* xrow = X + (size_t)(mt * 16 + (lane & 15)) * Dm;
  f32x8 c = {};
  for (int kt = 0; kt < Dm; kt += 32) {
    bf16x16 a = load_a_f32(xrow, kt, lane);
    bf16x16 b;
    const int kk0 = kt + ((lane >> 4) & 1) * 16;
#pragma unroll
    for (int i = 0; i < 16; ++i) b[i] = (bf16_t)W[(size_t)(kk0 + i) * (Hn * DH) + col];
    c = wmma_bf16(a, b, c);
  }
  const float bv = bias[col];
  const int h = col >> 6, dd = col & 63;
#pragma unroll
  for (int i = 0; i < 8; ++i) {
    const int row = mt * 16 + i + 8 * (lane >> 4);
    const int bb = row >> 10, ss = row & 1023;
    Y[(((size_t)bb * Hn + h) * Sq + ss) * DH + dd] = (bf16_t)(c[i] + bv);
  }
}

// ---------------------------------------------------------------------------
// Kernel 2: scores = (Q K^T)/8 + mask*-1e9, written to the weights buffer.
// ---------------------------------------------------------------------------
__global__ void scores_qk_kernel(const bf16_t* __restrict__ Qh,
                                 const bf16_t* __restrict__ Kh,
                                 const float* __restrict__ mask,
                                 float* __restrict__ wts) {
  const int lane = threadIdx.x & 31;
  const int k0 = blockIdx.x * 16, j0 = blockIdx.y * 16, bh = blockIdx.z;
  const int n = lane & 15;
  const bf16_t* qrow = Qh + ((size_t)bh * Sq + j0 + (lane & 15)) * DH;
  f32x8 c = {};
#pragma unroll
  for (int dstep = 0; dstep < 2; ++dstep) {
    bf16x16 a = load_a_bf16(qrow, dstep * 32, lane);
    const bf16_t* kp = Kh + ((size_t)bh * Sq + k0 + n) * DH + dstep * 32 +
                       ((lane >> 4) & 1) * 16;
    bf16x16 b = load_b_contig_bf16(kp);
    c = wmma_bf16(a, b, c);
  }
  const int bb = bh / Hn;
  const float mk = mask[(size_t)bb * Sq + k0 + n] * -1e9f;
#pragma unroll
  for (int i = 0; i < 8; ++i) {
    const int j = j0 + i + 8 * (lane >> 4);
    wts[((size_t)bh * Sq + j) * Sq + k0 + n] = c[i] * 0.125f + mk;
  }
}

// ---------------------------------------------------------------------------
// Kernel 3: scores += (q . pos)/8.  Batched over (b,h): per query j,
// GEMM [M=24 pad 32, K=64, N=1024]; B contiguous in d -> NT global loads.
// ---------------------------------------------------------------------------
__global__ void scores_pos_kernel(const bf16_t* __restrict__ Qh,
                                  const float* __restrict__ pos,
                                  float* __restrict__ wts) {
  const int lane = threadIdx.x & 31;
  const int k0 = blockIdx.x * 16;
  const int j = blockIdx.y;
  const int n = lane & 15, m = lane & 15;
  f32x8 c0 = {}, c1 = {};
#pragma unroll
  for (int dstep = 0; dstep < 2; ++dstep) {
    const int col0 = dstep * 32;
    const float* bp = pos + ((size_t)(k0 + n) * Sq + j) * DH + col0 +
                      ((lane >> 4) & 1) * 16;
    bf16x16 b = load_b_contig_f32_nt(bp);
    const bf16_t* ar0 = Qh + ((size_t)m * Sq + j) * DH;
    bf16x16 a0 = load_a_bf16(ar0, col0, lane);
    bf16x16 a1 = {};
    if (16 + m < BH) {
      const bf16_t* ar1 = Qh + ((size_t)(16 + m) * Sq + j) * DH;
      a1 = load_a_bf16(ar1, col0, lane);
    }
    c0 = wmma_bf16(a0, b, c0);
    c1 = wmma_bf16(a1, b, c1);
  }
#pragma unroll
  for (int i = 0; i < 8; ++i) {
    const int mm = i + 8 * (lane >> 4);
    wts[((size_t)mm * Sq + j) * Sq + k0 + n] += c0[i] * 0.125f;
    const int mh = 16 + mm;
    if (mh < BH) wts[((size_t)mh * Sq + j) * Sq + k0 + n] += c1[i] * 0.125f;
  }
}

// ---------------------------------------------------------------------------
// Kernel 4: row softmax over 1024 keys; one 256-thread block per row.
// ---------------------------------------------------------------------------
__global__ void softmax_kernel(float* __restrict__ wts) {
  __shared__ float red[256];
  float* p = wts + (size_t)blockIdx.x * Sq;
  const int t = threadIdx.x;
  float x0 = p[t], x1 = p[t + 256], x2 = p[t + 512], x3 = p[t + 768];
  float mx = fmaxf(fmaxf(x0, x1), fmaxf(x2, x3));
  red[t] = mx;
  __syncthreads();
  for (int s = 128; s > 0; s >>= 1) {
    if (t < s) red[t] = fmaxf(red[t], red[t + s]);
    __syncthreads();
  }
  mx = red[0];
  __syncthreads();
  x0 = __expf(x0 - mx); x1 = __expf(x1 - mx);
  x2 = __expf(x2 - mx); x3 = __expf(x3 - mx);
  red[t] = x0 + x1 + x2 + x3;
  __syncthreads();
  for (int s = 128; s > 0; s >>= 1) {
    if (t < s) red[t] += red[t + s];
    __syncthreads();
  }
  const float inv = 1.0f / red[0];
  p[t] = x0 * inv; p[t + 256] = x1 * inv;
  p[t + 512] = x2 * inv; p[t + 768] = x3 * inv;
}

// ---------------------------------------------------------------------------
// Kernel 5: ctx = weights(S x S) * V(S x DH) per (b,h).
// Block = 4 waves = 64 queries x 64 dh; each wave owns a distinct 16-query
// strip (A rows loaded exactly once), V tile async-staged into LDS (bf16
// byte copy -> GLOBAL_LOAD_ASYNC_TO_LDS_B128) and shared by all waves.
// ---------------------------------------------------------------------------
__global__ void ctx_v_kernel(const float* __restrict__ wts,
                             const bf16_t* __restrict__ Vh,
                             float* __restrict__ ctx) {
  __shared__ bf16_t Vt[32][64];
  const int bh = blockIdx.y, jbase = blockIdx.x * 64;
  const int t = threadIdx.x, lane = t & 31, wv = t >> 5;
  const int j0 = jbase + wv * 16;
  const float* wrow = wts + ((size_t)bh * Sq + j0 + (lane & 15)) * Sq;
  f32x8 c[4] = {{}, {}, {}, {}};
  for (int k0 = 0; k0 < Sq; k0 += 32) {
    __syncthreads();  // previous iteration's LDS reads done
    {
      const int r = t >> 2, cs = (t & 3) * 16;
      const bf16_t* src = Vh + ((size_t)bh * Sq + k0 + r) * DH + cs;
      async_ld_b128(&Vt[r][cs], src);
      async_ld_b128(&Vt[r][cs + 8], src + 8);
    }
    wait_async0();
    __syncthreads();  // tile visible to all waves
    bf16x16 a = load_a_f32(wrow, k0, lane);
    const int kh = ((lane >> 4) & 1) * 16;
#pragma unroll
    for (int dt = 0; dt < 4; ++dt) {
      bf16x16 b;
      const int n = dt * 16 + (lane & 15);
#pragma unroll
      for (int i = 0; i < 16; ++i) b[i] = Vt[kh + i][n];
      c[dt] = wmma_bf16(a, b, c[dt]);
    }
  }
  const int bb = bh / Hn, h = bh % Hn;
#pragma unroll
  for (int dt = 0; dt < 4; ++dt) {
    const int n = dt * 16 + (lane & 15);
#pragma unroll
    for (int i = 0; i < 8; ++i) {
      const int j = j0 + i + 8 * (lane >> 4);
      ctx[((size_t)bb * Sq + j) * Dm + h * DH + n] = c[dt][i];
    }
  }
}

// ---------------------------------------------------------------------------
// Kernel 6: ctx += weights . pos.  Per query j: GEMM [M=24 pad 32, K=1024,
// N=64]; pos slice pos[j,:,:] async-staged fp32 into LDS, converted to bf16
// at B-read; A rows (4 MB-strided weights) staged to bf16 manually.
// ---------------------------------------------------------------------------
__global__ void ctx_pos_kernel(const float* __restrict__ wts,
                               const float* __restrict__ pos,
                               float* __restrict__ ctx) {
  __shared__ bf16_t At[32][32];
  __shared__ float Bt[32][64];
  const int j = blockIdx.x;
  const int t = threadIdx.x, lane = t & 31, wv = t >> 5;
  const int d0 = wv * 16;
  f32x8 c0 = {}, c1 = {};
  for (int k0 = 0; k0 < Sq; k0 += 32) {
    __syncthreads();
    {
      const int r = t >> 2;
      const int cs16 = (t & 3) * 16;
      const float* sp = pos + ((size_t)j * Sq + k0 + r) * DH + cs16;
#pragma unroll
      for (int q = 0; q < 4; ++q) async_ld_b128(&Bt[r][cs16 + 4 * q], sp + 4 * q);
      if (k0 + 32 < Sq) __builtin_prefetch(sp + 32 * DH, 0, 1);
      const int cs8 = (t & 3) * 8;
      if (r < BH) {
        const float* src = wts + ((size_t)r * Sq + j) * Sq + k0 + cs8;
#pragma unroll
        for (int i = 0; i < 8; ++i) At[r][cs8 + i] = (bf16_t)src[i];
      } else {
#pragma unroll
        for (int i = 0; i < 8; ++i) At[r][cs8 + i] = (bf16_t)0.0f;
      }
    }
    wait_async0();
    __syncthreads();
    bf16x16 b;
    {
      const int n = d0 + (lane & 15);
      const int kh = ((lane >> 4) & 1) * 16;
#pragma unroll
      for (int i = 0; i < 16; ++i) b[i] = (bf16_t)Bt[kh + i][n];
    }
    const int kb = ((lane >> 4) & 1) * 16;
    const int m = lane & 15;
    bf16x16 a0, a1;
#pragma unroll
    for (int i = 0; i < 8; ++i) {
      a0[i] = At[m][kb + i];        a0[8 + i] = At[m][8 + kb + i];
      a1[i] = At[16 + m][kb + i];   a1[8 + i] = At[16 + m][8 + kb + i];
    }
    c0 = wmma_bf16(a0, b, c0);
    c1 = wmma_bf16(a1, b, c1);
  }
  const int n = d0 + (lane & 15);
#pragma unroll
  for (int i = 0; i < 8; ++i) {
    const int m = i + 8 * (lane >> 4);
    {
      const int bb = m / Hn, h = m % Hn;
      ctx[((size_t)bb * Sq + j) * Dm + h * DH + n] += c0[i];
    }
    const int mh = 16 + m;
    if (mh < BH) {
      const int bb = mh / Hn, h = mh % Hn;
      ctx[((size_t)bb * Sq + j) * Dm + h * DH + n] += c1[i];
    }
  }
}

// ---------------------------------------------------------------------------
// Kernel 7: out = ctx(2048x768) * Wo(768x768) + bo (fp32 out, plain layout).
// ---------------------------------------------------------------------------
__global__ void out_proj_kernel(const float* __restrict__ Xc,
                                const float* __restrict__ Wo,
                                const float* __restrict__ bo,
                                float* __restrict__ out) {
  const int lane = threadIdx.x & 31;
  const int nt = blockIdx.x, mt = blockIdx.y;
  const int col = nt * 16 + (lane & 15);
  const float* xrow = Xc + (size_t)(mt * 16 + (lane & 15)) * Dm;
  f32x8 c = {};
  for (int kt = 0; kt < Dm; kt += 32) {
    bf16x16 a = load_a_f32(xrow, kt, lane);
    bf16x16 b;
    const int kk0 = kt + ((lane >> 4) & 1) * 16;
#pragma unroll
    for (int i = 0; i < 16; ++i) b[i] = (bf16_t)Wo[(size_t)(kk0 + i) * Dm + col];
    c = wmma_bf16(a, b, c);
  }
  const float bv = bo[col];
#pragma unroll
  for (int i = 0; i < 8; ++i) {
    const int row = mt * 16 + i + 8 * (lane >> 4);
    out[(size_t)row * Dm + col] = c[i] + bv;
  }
}

// ---------------------------------------------------------------------------
extern "C" void kernel_launch(void* const* d_in, const int* in_sizes, int n_in,
                              void* d_out, int out_size, void* d_ws,
                              size_t ws_size, hipStream_t stream) {
  const float* q_in  = (const float*)d_in[0];
  const float* k_in  = (const float*)d_in[1];
  const float* v_in  = (const float*)d_in[2];
  const float* pos   = (const float*)d_in[3];
  const float* mask  = (const float*)d_in[4];
  const float* Wq    = (const float*)d_in[5];
  const float* bq    = (const float*)d_in[6];
  const float* Wk    = (const float*)d_in[7];
  const float* bk    = (const float*)d_in[8];
  const float* Wv    = (const float*)d_in[9];
  const float* bv    = (const float*)d_in[10];
  const float* Wo    = (const float*)d_in[11];
  const float* bo    = (const float*)d_in[12];

  // Workspace: bf16 Q/K/V in head-major layout + fp32 ctx.  ~15.7 MB total.
  const size_t NQ = (size_t)BH * Sq * DH;  // 1,572,864 elements
  bf16_t* qh = (bf16_t*)d_ws;
  bf16_t* kh = qh + NQ;
  bf16_t* vh = kh + NQ;
  float* ctx = (float*)(vh + NQ);

  float* out = (float*)d_out;                       // (B, S, D)
  float* wts = out + (size_t)Bc * Sq * Dm;          // (B, H, S, S)

  // 1) QKV projections (fp32 -> bf16 head-major)
  dim3 gproj(Dm / 16, (Bc * Sq) / 16);
  proj_kernel<<<gproj, 32, 0, stream>>>(q_in, Wq, bq, qh);
  proj_kernel<<<gproj, 32, 0, stream>>>(k_in, Wk, bk, kh);
  proj_kernel<<<gproj, 32, 0, stream>>>(v_in, Wv, bv, vh);

  // 2) QK^T / 8 + mask  -> weights buffer
  scores_qk_kernel<<<dim3(Sq / 16, Sq / 16, BH), 32, 0, stream>>>(qh, kh, mask,
                                                                  wts);
  // 3) += (q . pos)/8  (batched over bh, per query j)
  scores_pos_kernel<<<dim3(Sq / 16, Sq), 32, 0, stream>>>(qh, pos, wts);

  // 4) row softmax (mask already applied)
  softmax_kernel<<<BH * Sq, 256, 0, stream>>>(wts);

  // 5) ctx = weights * V  (async LDS staging, shared V tile)
  ctx_v_kernel<<<dim3(Sq / 64, BH), 128, 0, stream>>>(wts, vh, ctx);

  // 6) ctx += weights . pos  (async LDS staging of pos slice)
  ctx_pos_kernel<<<Sq, 128, 0, stream>>>(wts, pos, ctx);

  // 7) out = ctx * Wo + bo
  out_proj_kernel<<<gproj, 32, 0, stream>>>(ctx, Wo, bo, out);
}
